// SpatialPolicyHead_2422361555019
// MI455X (gfx1250) — matrix operand order repaired and checked by reading, not compiled, not run on hardware
//
#include <hip/hip_runtime.h>
#include <hip/hip_bf16.h>

// ---------------------------------------------------------------------------
// Shapes (compile-time constants from the reference)
// ---------------------------------------------------------------------------
#define BB   256          // batch
#define SS   128          // seq len
#define HH   1024         // hidden
#define DD   256          // proj dim
#define VV   1858         // moves
#define MM   (BB * 64)    // 16384 GEMM rows (64 squares per batch)

#define LDS_STRIDE 40     // padded row stride in bf16 units (80B: 16B-aligned,
                          // 20-dword bank stride -> conflict-free across 16 lanes)

#if __has_builtin(__builtin_amdgcn_sched_barrier)
#define SCHED_FENCE() __builtin_amdgcn_sched_barrier(0)
#else
#define SCHED_FENCE()
#endif

typedef __attribute__((ext_vector_type(16))) __bf16 v16bf;
typedef __attribute__((ext_vector_type(8)))  float  v8f;

union FragBF {
    v16bf v;
    unsigned int u[8];
};

__device__ __forceinline__ unsigned short f2bf_rne(float f) {
    unsigned int u = __float_as_uint(f);
    u += 0x7FFFu + ((u >> 16) & 1u);
    return (unsigned short)(u >> 16);
}
__device__ __forceinline__ unsigned int pk_bf16(float a, float b) {
    return (unsigned int)f2bf_rne(a) | ((unsigned int)f2bf_rne(b) << 16);
}

// ---------------------------------------------------------------------------
// Kernel 1: convert Wf / Wt (D x H fp32) -> bf16 in workspace
// ---------------------------------------------------------------------------
__global__ void cvt_w_kernel(const float* __restrict__ Wf,
                             const float* __restrict__ Wt,
                             unsigned short* __restrict__ wfb,
                             unsigned short* __restrict__ wtb) {
    int i = blockIdx.x * 256 + threadIdx.x;       // 0 .. D*H-1 (262144)
    wfb[i] = f2bf_rne(Wf[i]);
    wtb[i] = f2bf_rne(Wt[i]);
}

// ---------------------------------------------------------------------------
// Kernel 2: fused fp/tp projection GEMM via v_wmma_f32_16x16x32_bf16.
//   LDS-staged, double-buffered software pipeline:
//     step s: regs -> LDS[st], barrier, issue global loads for s+1 (pinned
//     above compute by sched_barrier), then ds_load fragments two-deep so
//     every WMMA's operands are already resident when it issues.
//   Block = 256 threads = 8 waves; wave (wm 0..3, wn 0..1):
//     rows m0 = bx*64 + wm*16, cols n0 = by*128 + wn*64 (+ j*16, j=0..3)
// ---------------------------------------------------------------------------
__global__ void __launch_bounds__(256)
proj_gemm_kernel(const float* __restrict__ hid,
                 const unsigned short* __restrict__ wfb,
                 const unsigned short* __restrict__ wtb,
                 const float* __restrict__ bf,
                 const float* __restrict__ bt,
                 float* __restrict__ fpo,
                 float* __restrict__ tpo) {
    __shared__ unsigned short ldsA[2][64 * LDS_STRIDE];        // 2 x 5120 B
    __shared__ unsigned short ldsB[2][2][128 * LDS_STRIDE];    // 2 x 2 x 10240 B

    const int tid  = threadIdx.x;
    const int lane = tid & 31;
    const int wid  = tid >> 5;
    const int wm   = wid & 3;
    const int wn   = wid >> 2;

    // ---------------- fill-side indices ----------------
    const int fr  = tid >> 2;                      // A row 0..63
    const int fc8 = (tid & 3) << 3;                // A col chunk 0/8/16/24
    const int amrow = blockIdx.x * 64 + fr;
    const long afill = ((long)((amrow >> 6) * SS + 3 + (amrow & 63))) * HH + fc8;

    const int bmat = tid >> 7;                     // 0: Wf, 1: Wt (wave-uniform)
    const int bn   = tid & 127;
    const unsigned short* __restrict__ wsel = bmat ? wtb : wfb;
    const long bfill = (long)(blockIdx.y * 128 + bn) * HH;

    // ---------------- compute-side indices ----------------
    const int hiK  = lane >> 4;                    // 0: lanes 0-15, 1: lanes 16-31
    const unsigned aIdx = (unsigned)((wm * 16 + (lane & 15)) * LDS_STRIDE + hiK * 16);
    unsigned bIdx[4];
    int      ncol[4];
#pragma unroll
    for (int j = 0; j < 4; ++j) {
        const int nl = wn * 64 + j * 16 + (lane & 15);
        bIdx[j] = (unsigned)(nl * LDS_STRIDE + hiK * 16);
        ncol[j] = blockIdx.y * 128 + nl;
    }

    v8f accf[4] = {};
    v8f acct[4] = {};

    // ---------------- prologue: issue loads for step 0 ----------------
    float4 a0 = *reinterpret_cast<const float4*>(hid + afill);
    float4 a1 = *reinterpret_cast<const float4*>(hid + afill + 4);
    const uint4* bp = reinterpret_cast<const uint4*>(wsel + bfill);
    uint4 b0 = bp[0], b1 = bp[1], b2 = bp[2], b3 = bp[3];

    for (int s = 0; s < HH / 32; ++s) {
        const int st = s & 1;

        // ---- regs -> LDS (convert A to bf16 here, once per block)
        {
            uint4 q;
            q.x = pk_bf16(a0.x, a0.y);
            q.y = pk_bf16(a0.z, a0.w);
            q.z = pk_bf16(a1.x, a1.y);
            q.w = pk_bf16(a1.z, a1.w);
            *reinterpret_cast<uint4*>(&ldsA[st][fr * LDS_STRIDE + fc8]) = q;

            uint4* d = reinterpret_cast<uint4*>(&ldsB[st][bmat][bn * LDS_STRIDE]);
            d[0] = b0; d[1] = b1; d[2] = b2; d[3] = b3;
        }
        __syncthreads();

        // ---- issue global loads for step s+1; the sched fence keeps them
        //      above the WMMA block so their latency hides behind compute.
        if (s < HH / 32 - 1) {
            const long ka = afill + (long)(s + 1) * 32;
            a0 = *reinterpret_cast<const float4*>(hid + ka);
            a1 = *reinterpret_cast<const float4*>(hid + ka + 4);
            const uint4* bq =
                reinterpret_cast<const uint4*>(wsel + bfill + (s + 1) * 32);
            b0 = bq[0]; b1 = bq[1]; b2 = bq[2]; b3 = bq[3];
        }
        SCHED_FENCE();

        // ---- A fragment from LDS (contiguous k order, same as B):
        //   lanes 0-15 : K = {0..15};  lanes 16-31: K = {16..31}
        FragBF a;
        {
            uint4 lo = *reinterpret_cast<const uint4*>(&ldsA[st][aIdx]);
            uint4 hi = *reinterpret_cast<const uint4*>(&ldsA[st][aIdx + 8]);
            a.u[0] = lo.x; a.u[1] = lo.y; a.u[2] = lo.z; a.u[3] = lo.w;
            a.u[4] = hi.x; a.u[5] = hi.y; a.u[6] = hi.z; a.u[7] = hi.w;
        }

        // ---- B fragments, two-deep: while WMMA-ing j, loads for j+1 fly.
        FragBF fb, tb;
        {
            const uint4* pf = reinterpret_cast<const uint4*>(&ldsB[st][0][bIdx[0]]);
            const uint4* pt = reinterpret_cast<const uint4*>(&ldsB[st][1][bIdx[0]]);
            uint4 f0 = pf[0], f1 = pf[1], t0 = pt[0], t1 = pt[1];
            fb.u[0] = f0.x; fb.u[1] = f0.y; fb.u[2] = f0.z; fb.u[3] = f0.w;
            fb.u[4] = f1.x; fb.u[5] = f1.y; fb.u[6] = f1.z; fb.u[7] = f1.w;
            tb.u[0] = t0.x; tb.u[1] = t0.y; tb.u[2] = t0.z; tb.u[3] = t0.w;
            tb.u[4] = t1.x; tb.u[5] = t1.y; tb.u[6] = t1.z; tb.u[7] = t1.w;
        }
#pragma unroll
        for (int j = 0; j < 4; ++j) {
            FragBF fbn, tbn;
            if (j < 3) {
                const uint4* pf =
                    reinterpret_cast<const uint4*>(&ldsB[st][0][bIdx[j + 1]]);
                const uint4* pt =
                    reinterpret_cast<const uint4*>(&ldsB[st][1][bIdx[j + 1]]);
                uint4 f0 = pf[0], f1 = pf[1], t0 = pt[0], t1 = pt[1];
                fbn.u[0] = f0.x; fbn.u[1] = f0.y; fbn.u[2] = f0.z; fbn.u[3] = f0.w;
                fbn.u[4] = f1.x; fbn.u[5] = f1.y; fbn.u[6] = f1.z; fbn.u[7] = f1.w;
                tbn.u[0] = t0.x; tbn.u[1] = t0.y; tbn.u[2] = t0.z; tbn.u[3] = t0.w;
                tbn.u[4] = t1.x; tbn.u[5] = t1.y; tbn.u[6] = t1.z; tbn.u[7] = t1.w;
            }
            accf[j] = __builtin_amdgcn_wmma_f32_16x16x32_bf16(
                false, a.v, false, fb.v, (short)0, accf[j], false, false);
            acct[j] = __builtin_amdgcn_wmma_f32_16x16x32_bf16(
                false, a.v, false, tb.v, (short)0, acct[j], false, false);
            if (j < 3) { fb = fbn; tb = tbn; }
        }
        // One barrier per step suffices with double buffering: the WAR on
        // buffer st (rewritten at step s+2) is ordered by the s+1 barrier.
    }

    // ---- Store with bias. C/D layout: VGPR r, lanes 0-15 -> M=r, lanes 16-31 -> M=r+8
    const int rbase = blockIdx.x * 64 + wm * 16 + hiK * 8;
#pragma unroll
    for (int j = 0; j < 4; ++j) {
        const int   n   = ncol[j];
        const float bfv = bf[n];
        const float btv = bt[n];
#pragma unroll
        for (int r = 0; r < 8; ++r) {
            const long o = (long)(rbase + r) * DD + n;
            fpo[o] = accf[j][r] + bfv;
            tpo[o] = acct[j][r] + btv;
        }
    }
}

// ---------------------------------------------------------------------------
// Kernel 3: gp[b,d] = dot(hidden[b,0,:], Wg[d,:]) + bg[d]  (fp32 exact, tiny)
// ---------------------------------------------------------------------------
__global__ void gp_gemm_kernel(const float* __restrict__ hid,
                               const float* __restrict__ Wg,
                               const float* __restrict__ bg,
                               float* __restrict__ gp) {
    __shared__ float4 hrow[HH / 4];
    const int b = blockIdx.x;
    const int d = threadIdx.x;

    const float4* h4 = reinterpret_cast<const float4*>(hid + (long)b * SS * HH);
    hrow[d] = h4[d];                    // 256 threads * 16B = 4KB = full row 0
    __syncthreads();

    const float4* w4 = reinterpret_cast<const float4*>(Wg + (long)d * HH);
    float acc = 0.f;
#pragma unroll 4
    for (int i = 0; i < HH / 4; ++i) {
        float4 h = hrow[i];
        float4 w = w4[i];
        acc += h.x * w.x + h.y * w.y + h.z * w.z + h.w * w.w;
    }
    gp[(long)b * DD + d] = acc + bg[d];
}

// ---------------------------------------------------------------------------
// Kernel 4: out[b,v] = sum_d relu(fp[b,from[v],d]*tp[b,to[v],d] + gp[b,d]
//                                 + promo[pt[v],d]) * Ws[d] + bs
//   One wave32 per (b,v); 8 waves per block.
// ---------------------------------------------------------------------------
__global__ void policy_out_kernel(const float* __restrict__ fp,
                                  const float* __restrict__ tp,
                                  const float* __restrict__ gp,
                                  const float* __restrict__ promo,
                                  const float* __restrict__ Ws,
                                  const float* __restrict__ bs,
                                  const int* __restrict__ fromv,
                                  const int* __restrict__ tov,
                                  const int* __restrict__ ptv,
                                  float* __restrict__ out) {
    const int lane = threadIdx.x & 31;
    const int wid  = threadIdx.x >> 5;
    const unsigned idx = blockIdx.x * 8u + (unsigned)wid;   // 0 .. B*V-1
    const unsigned b = idx / VV;
    const unsigned v = idx % VV;

    const float4* f4 = reinterpret_cast<const float4*>(fp + ((long)(b * 64 + fromv[v])) * DD);
    const float4* t4 = reinterpret_cast<const float4*>(tp + ((long)(b * 64 + tov[v])) * DD);
    const float4* g4 = reinterpret_cast<const float4*>(gp + (long)b * DD);
    const float4* p4 = reinterpret_cast<const float4*>(promo + (long)ptv[v] * DD);
    const float4* w4 = reinterpret_cast<const float4*>(Ws);

    float s = 0.f;
#pragma unroll
    for (int i = lane; i < DD / 4; i += 32) {      // 2 iterations
        float4 f = f4[i], t = t4[i], g = g4[i], p = p4[i], w = w4[i];
        float c0 = fmaxf(f.x * t.x + g.x + p.x, 0.f);
        float c1 = fmaxf(f.y * t.y + g.y + p.y, 0.f);
        float c2 = fmaxf(f.z * t.z + g.z + p.z, 0.f);
        float c3 = fmaxf(f.w * t.w + g.w + p.w, 0.f);
        s += c0 * w.x + c1 * w.y + c2 * w.z + c3 * w.w;
    }
#pragma unroll
    for (int off = 16; off > 0; off >>= 1)
        s += __shfl_xor(s, off, 32);
    if (lane == 0)
        out[idx] = s + bs[0];
}

// ---------------------------------------------------------------------------
// Launch
// ---------------------------------------------------------------------------
extern "C" void kernel_launch(void* const* d_in, const int* in_sizes, int n_in,
                              void* d_out, int out_size, void* d_ws, size_t ws_size,
                              hipStream_t stream) {
    const float* hid   = (const float*)d_in[0];
    const float* Wf    = (const float*)d_in[1];
    const float* bf    = (const float*)d_in[2];
    const float* Wt    = (const float*)d_in[3];
    const float* bt    = (const float*)d_in[4];
    const float* Wg    = (const float*)d_in[5];
    const float* bg    = (const float*)d_in[6];
    const float* promo = (const float*)d_in[7];
    const float* Ws    = (const float*)d_in[8];
    const float* bs    = (const float*)d_in[9];
    const int*   fromv = (const int*)d_in[10];
    const int*   tov   = (const int*)d_in[11];
    const int*   ptv   = (const int*)d_in[12];
    float*       out   = (float*)d_out;

    char* ws = (char*)d_ws;
    unsigned short* wfb = (unsigned short*)ws;                          // 512 KB
    unsigned short* wtb = (unsigned short*)(ws + (512u << 10));         // 512 KB
    float* fp = (float*)(ws + (1u << 20));                              // 16 MB
    float* tp = (float*)(ws + (1u << 20) + (16u << 20));                // 16 MB
    float* gp = (float*)(ws + (1u << 20) + (32u << 20));                // 256 KB

    cvt_w_kernel<<<(DD * HH) / 256, 256, 0, stream>>>(Wf, Wt, wfb, wtb);
    gp_gemm_kernel<<<BB, DD, 0, stream>>>(hid, Wg, bg, gp);
    proj_gemm_kernel<<<dim3(MM / 64, DD / 128), 256, 0, stream>>>(
        hid, wfb, wtb, bf, bt, fp, tp);
    policy_out_kernel<<<(BB * VV) / 8, 256, 0, stream>>>(
        fp, tp, gp, promo, Ws, bs, fromv, tov, ptv, out);
}
